// CICDM_Net_1640677507714
// MI455X (gfx1250) — matrix operand order repaired and checked by reading, not compiled, not run on hardware
//
#include <hip/hip_runtime.h>
#include <hip/hip_bf16.h>
#include <stdint.h>

#define B_STU 512
#define L_LOG 128
#define C_CONC 512
#define E_EXER 10000
#define P_POT 32

typedef __attribute__((ext_vector_type(16))) _Float16 v16h;
typedef __attribute__((ext_vector_type(8)))  _Float16 v8h;
typedef __attribute__((ext_vector_type(8)))  float    v8f;

__device__ __forceinline__ float sigmoidf(float x) {
    return 1.0f / (1.0f + __expf(-x));
}

// ---- WMMA fragment loaders (16x16x32 f16, wave32, contiguous k = 16h + s order) ----
// A (16x32, row-major source with stride lda halves):
//   lanes 0-15  (M=lane):    halves = K[kb+0..15]
//   lanes 16-31 (M=lane-16): halves = K[kb+16..31]
__device__ __forceinline__ v16h load_a_frag(const _Float16* base, int lda, int lane, int kbase) {
    int m    = lane & 15;
    int koff = kbase + ((lane >> 4) << 4);
    const _Float16* p = base + (size_t)m * (size_t)lda + koff;
    v8h lo = *(const v8h*)(p);
    v8h hi = *(const v8h*)(p + 8);
    v16h a;
#pragma unroll
    for (int i = 0; i < 8; ++i) { a[i] = lo[i]; a[i + 8] = hi[i]; }
    return a;
}
// B (32x16): lane n=lane&15 holds column n; lanes 0-15 hold K[kb+0..15],
// lanes 16-31 hold K[kb+16..31] -> one contiguous v16h per lane from a
// K-contiguous (i.e. transposed / row-per-column) source.
__device__ __forceinline__ v16h load_b_frag_contig(const _Float16* colbase, int kbase, int lane) {
    int ksel = (lane < 16) ? 0 : 16;
    return *(const v16h*)(colbase + kbase + ksel);
}

// ---------------- K0: W = sigmoid(ecw)*adj ; Wf32 ; W2h = W/rowsum ----------
__global__ void k0_prep_w(const float* __restrict__ ecw, const float* __restrict__ adj,
                          float* __restrict__ Wf, _Float16* __restrict__ W2h) {
    int e   = blockIdx.x;
    int tid = threadIdx.x;
    size_t row = (size_t)e * C_CONC;
    int c0 = tid, c1 = tid + 256;
    float w0 = sigmoidf(ecw[row + c0]) * adj[row + c0];
    float w1 = sigmoidf(ecw[row + c1]) * adj[row + c1];
    Wf[row + c0] = w0;
    Wf[row + c1] = w1;
    __shared__ float red[256];
    red[tid] = w0 + w1;
    __syncthreads();
#pragma unroll
    for (int s = 128; s > 0; s >>= 1) {
        if (tid < s) red[tid] += red[tid + s];
        __syncthreads();
    }
    float inv = 1.0f / red[0];   // >=1 forced tag per row -> rowsum > 0
    W2h[row + c0] = (_Float16)(w0 * inv);
    W2h[row + c1] = (_Float16)(w1 * inv);
}

// ---------------- K1: ragged-log gather -> A1h, validh ----------------------
__global__ void k1_a1(const float* __restrict__ Wf, const int* __restrict__ ids,
                      const float* __restrict__ scores, const unsigned char* __restrict__ mask,
                      _Float16* __restrict__ A1h, _Float16* __restrict__ validh) {
    int b   = blockIdx.x;
    int tid = threadIdx.x;
    __shared__ int   s_id[L_LOG];
    __shared__ float s_sc[L_LOG];
    __shared__ int   s_mk[L_LOG];
    if (tid < L_LOG) {
        s_id[tid] = ids[b * L_LOG + tid];
        s_sc[tid] = scores[b * L_LOG + tid];
        s_mk[tid] = (int)mask[b * L_LOG + tid];
    }
    __syncthreads();
    int c0 = tid, c1 = tid + 256;
    float wsum0 = 0.f, wsum1 = 0.f, xw0 = 0.f, xw1 = 0.f;
    for (int l = 0; l < L_LOG; ++l) {
        if (s_mk[l]) {
            size_t row = (size_t)s_id[l] * C_CONC;
            float s = s_sc[l];
            float w0 = Wf[row + c0], w1 = Wf[row + c1];
            wsum0 += w0; xw0 += s * w0;
            wsum1 += w1; xw1 += s * w1;
        }
    }
    size_t o = (size_t)b * C_CONC;
    A1h[o + c0]    = (_Float16)(wsum0 > 0.f ? xw0 / wsum0 : 0.f);
    A1h[o + c1]    = (_Float16)(wsum1 > 0.f ? xw1 / wsum1 : 0.f);
    validh[o + c0] = (_Float16)(wsum0 > 0.f ? 1.f : 0.f);
    validh[o + c1] = (_Float16)(wsum1 > 0.f ? 1.f : 0.f);
}

// ---------------- K2: Bmat = masked softmax over L of gathered pote rows ----
__global__ void k2_bmat(const float* __restrict__ epw, const int* __restrict__ ids,
                        const float* __restrict__ scores, const unsigned char* __restrict__ mask,
                        _Float16* __restrict__ Bmath) {
    int b = blockIdx.x;
    int p = threadIdx.x;  // 32 threads
    float num = 0.f, den = 0.f;
    for (int l = 0; l < L_LOG; ++l) {
        if (mask[b * L_LOG + l]) {
            int e = ids[b * L_LOG + l];
            float ev = __expf(epw[(size_t)e * P_POT + p]);
            den += ev;
            num += scores[b * L_LOG + l] * ev;
        }
    }
    Bmath[b * P_POT + p] = (_Float16)(den > 0.f ? num / den : 0.f);
}

// ---------------- K3a: column max of conc_conc_w ----------------------------
__global__ void k3a_colmax(const float* __restrict__ ccw, float* __restrict__ colmax) {
    int d = blockIdx.x;
    int t = threadIdx.x;  // 32 threads
    float mx = -3.4e38f;
    for (int c = t; c < C_CONC; c += 32)
        mx = fmaxf(mx, ccw[(size_t)c * C_CONC + d]);
#pragma unroll
    for (int s = 16; s > 0; s >>= 1)
        mx = fmaxf(mx, __shfl_xor(mx, s, 32));
    if (t == 0) colmax[d] = mx;
}

// ---------------- K3b: expWhT[n*C+k] = exp(ccw[k,n] - colmax[n]) as f16 -----
__global__ void k3b_expw(const float* __restrict__ ccw, const float* __restrict__ colmax,
                         _Float16* __restrict__ expWhT) {
    int idx = blockIdx.x * blockDim.x + threadIdx.x;  // C*C threads
    int n = idx >> 9, k = idx & (C_CONC - 1);
    expWhT[idx] = (_Float16)__expf(ccw[(size_t)k * C_CONC + n] - colmax[n]);
}

// ---------------- K4: D2 = row softmax of exer_pote_w -----------------------
__global__ void k4_d2(const float* __restrict__ epw, _Float16* __restrict__ D2h) {
    int e = blockIdx.x;
    int t = threadIdx.x;  // 32 threads
    float v = epw[(size_t)e * P_POT + t];
    float mx = v;
#pragma unroll
    for (int s = 16; s > 0; s >>= 1) mx = fmaxf(mx, __shfl_xor(mx, s, 32));
    float ev = __expf(v - mx);
    float sum = ev;
#pragma unroll
    for (int s = 16; s > 0; s >>= 1) sum += __shfl_xor(sum, s, 32);
    D2h[e * P_POT + t] = (_Float16)(ev / sum);
}

// ---------------- K5: WMMA  num=A1@expW, den=valid@expW, A=num/den ----------
__global__ void k5_gemm_a(const _Float16* __restrict__ A1h, const _Float16* __restrict__ validh,
                          const _Float16* __restrict__ expWhT,
                          float* __restrict__ Aout, _Float16* __restrict__ Ah) {
    int wave = threadIdx.x >> 5;
    int lane = threadIdx.x & 31;
    int tile = blockIdx.x * 8 + wave;           // 32x32 = 1024 tiles, 128 blocks
    int mt = tile >> 5, nt = tile & 31;
    int ncol = nt * 16 + (lane & 15);
    const _Float16* bcol = expWhT + (size_t)ncol * C_CONC;
    const _Float16* arow = A1h    + (size_t)(mt * 16) * C_CONC;
    const _Float16* vrow = validh + (size_t)(mt * 16) * C_CONC;
    v8f acc_num = {}; v8f acc_den = {};
#pragma unroll 4
    for (int kb = 0; kb < C_CONC; kb += 32) {
        v16h a  = load_a_frag(arow, C_CONC, lane, kb);
        v16h av = load_a_frag(vrow, C_CONC, lane, kb);
        v16h bf = load_b_frag_contig(bcol, kb, lane);
        acc_num = __builtin_amdgcn_wmma_f32_16x16x32_f16(false, a,  false, bf, (short)0, acc_num, false, false);
        acc_den = __builtin_amdgcn_wmma_f32_16x16x32_f16(false, av, false, bf, (short)0, acc_den, false, false);
    }
#pragma unroll
    for (int v = 0; v < 8; ++v) {
        int row = mt * 16 + v + ((lane >> 4) << 3);
        int col = nt * 16 + (lane & 15);
        float d = acc_den[v];
        float a = (d > 0.f) ? acc_num[v] / d : 0.f;
        Aout[(size_t)row * C_CONC + col] = a;
        Ah[(size_t)row * C_CONC + col]   = (_Float16)a;
    }
}

// ---------------- K6: WMMA  Y = combine(A@W2^T, Bmat@D2^T) ------------------
__global__ void k6_gemm_y(const _Float16* __restrict__ Ah, const _Float16* __restrict__ W2h,
                          const _Float16* __restrict__ Bmath, const _Float16* __restrict__ D2h,
                          const float* __restrict__ lambd, const float* __restrict__ guess,
                          const float* __restrict__ slide, float* __restrict__ Yout) {
    int wave = threadIdx.x >> 5;
    int lane = threadIdx.x & 31;
    int tile = blockIdx.x * 8 + wave;           // 32 * 625 = 20000 tiles, 2500 blocks
    int mt = tile / 625, nt = tile - mt * 625;
    int ecol = nt * 16 + (lane & 15);
    const _Float16* bcol = W2h + (size_t)ecol * C_CONC;   // B[k][n] = W2[e_n][k]
    const _Float16* arow = Ah  + (size_t)(mt * 16) * C_CONC;
    v8f acc = {};
#pragma unroll 4
    for (int kb = 0; kb < C_CONC; kb += 32) {
        v16h a  = load_a_frag(arow, C_CONC, lane, kb);
        v16h bf = load_b_frag_contig(bcol, kb, lane);
        acc = __builtin_amdgcn_wmma_f32_16x16x32_f16(false, a, false, bf, (short)0, acc, false, false);
    }
    // Y_B tile: K = P = 32 in a single WMMA
    v16h a2 = load_a_frag(Bmath + (size_t)(mt * 16) * P_POT, P_POT, lane, 0);
    v16h b2 = load_b_frag_contig(D2h + (size_t)ecol * P_POT, 0, lane);
    v8f zero = {};
    v8f accB = __builtin_amdgcn_wmma_f32_16x16x32_f16(false, a2, false, b2, (short)0, zero, false, false);

    float lam = sigmoidf(lambd[ecol]);
    float sl  = sigmoidf(slide[ecol]);
    float gu  = sigmoidf(guess[ecol]);
#pragma unroll
    for (int v = 0; v < 8; ++v) {
        int b = mt * 16 + v + ((lane >> 4) << 3);
        float y_ = (1.0f - lam) * acc[v] + lam * accB[v];
        y_ = fminf(fmaxf(y_, 1e-8f), 1.0f - 1e-8f);
        float y = (1.0f - sl) * y_ + gu * (1.0f - y_);
        Yout[(size_t)b * E_EXER + ecol] = y;
    }
}

extern "C" void kernel_launch(void* const* d_in, const int* in_sizes, int n_in,
                              void* d_out, int out_size, void* d_ws, size_t ws_size,
                              hipStream_t stream) {
    const float* scores = (const float*)d_in[0];
    const float* adj    = (const float*)d_in[1];
    const float* ecw    = (const float*)d_in[2];
    const float* ccw    = (const float*)d_in[3];
    const float* epw    = (const float*)d_in[4];
    const float* lambd  = (const float*)d_in[5];
    const float* guess  = (const float*)d_in[6];
    const float* slide  = (const float*)d_in[7];
    const unsigned char* mask = (const unsigned char*)d_in[8];
    const int* ids      = (const int*)d_in[9];
    float* out = (float*)d_out;
    float* Aout = out;                          // [B,C]
    float* Yout = out + (size_t)B_STU * C_CONC; // [B,E]

    char* ws = (char*)d_ws;
    size_t off = 0;
    float*    Wf     = (float*)(ws + off);    off += (size_t)E_EXER * C_CONC * 4;  // 20.48 MB
    _Float16* W2h    = (_Float16*)(ws + off); off += (size_t)E_EXER * C_CONC * 2;  // 10.24 MB
    _Float16* expWhT = (_Float16*)(ws + off); off += (size_t)C_CONC * C_CONC * 2;
    _Float16* A1h    = (_Float16*)(ws + off); off += (size_t)B_STU * C_CONC * 2;
    _Float16* validh = (_Float16*)(ws + off); off += (size_t)B_STU * C_CONC * 2;
    _Float16* Ah     = (_Float16*)(ws + off); off += (size_t)B_STU * C_CONC * 2;
    _Float16* Bmath  = (_Float16*)(ws + off); off += (size_t)B_STU * P_POT * 2;
    _Float16* D2h    = (_Float16*)(ws + off); off += (size_t)E_EXER * P_POT * 2;
    float*    colmax = (float*)(ws + off);    off += (size_t)C_CONC * 4;

    hipLaunchKernelGGL(k0_prep_w, dim3(E_EXER), dim3(256), 0, stream, ecw, adj, Wf, W2h);
    hipLaunchKernelGGL(k1_a1, dim3(B_STU), dim3(256), 0, stream, Wf, ids, scores, mask, A1h, validh);
    hipLaunchKernelGGL(k2_bmat, dim3(B_STU), dim3(32), 0, stream, epw, ids, scores, mask, Bmath);
    hipLaunchKernelGGL(k3a_colmax, dim3(C_CONC), dim3(32), 0, stream, ccw, colmax);
    hipLaunchKernelGGL(k3b_expw, dim3((C_CONC * C_CONC) / 256), dim3(256), 0, stream, ccw, colmax, expWhT);
    hipLaunchKernelGGL(k4_d2, dim3(E_EXER), dim3(32), 0, stream, epw, D2h);
    hipLaunchKernelGGL(k5_gemm_a, dim3(128), dim3(256), 0, stream, A1h, validh, expWhT, Aout, Ah);
    hipLaunchKernelGGL(k6_gemm_y, dim3(2500), dim3(256), 0, stream, Ah, W2h, Bmath, D2h,
                       lambd, guess, slide, Yout);
}
